// WarlightPolicyNet_8813272891627
// MI455X (gfx1250) — matrix operand order repaired and checked by reading, not compiled, not run on hardware
//
#include <hip/hip_runtime.h>
#include <hip/hip_bf16.h>

typedef __attribute__((ext_vector_type(16))) _Float16 v16h;
typedef __attribute__((ext_vector_type(8)))  _Float16 v8h;
typedef __attribute__((ext_vector_type(8)))  float    v8f;

union V16U { v16h v; v8h h[2]; };

// ---------------------------------------------------------------------------
// WMMA helpers (CDNA5 wave32, 16x16x32 f16 -> f32)
// ---------------------------------------------------------------------------
__device__ inline v8f wmma16(v16h a, v16h b, v8f c) {
  return __builtin_amdgcn_wmma_f32_16x16x32_f16(false, a, false, b, (short)0, c, false, false);
}

// A-matrix fragment from a row-major f16 tile (pitch in halves, 16B aligned).
// 16-bit A 16x32 layout: lane l holds row M=l&15; K = kt*32 + (l>>4)*8 + {0..7}
// in halves[0..7] and +16 in halves[8..15].
__device__ inline v16h load_a_frag(const _Float16* tile, int lane, int kt, int pitch) {
  int row  = lane & 15;
  int half = lane >> 4;
  const _Float16* p = tile + row * pitch + kt * 32 + half * 8;
  V16U u;
  u.h[0] = *(const v8h*)(p);
  u.h[1] = *(const v8h*)(p + 16);
  return u.v;
}

// B-matrix fragment from pre-swizzled pack: frag is 512 halves; lane l holds
// column N=l&15, K = (l>>4)*8 + (i&7) + (i>>3)*16  ->  contiguous 16 halves per lane.
__device__ inline v16h load_b_frag(const _Float16* frag, int lane) {
  const _Float16* p = frag + lane * 16;
  V16U u;
  u.h[0] = *(const v8h*)(p);
  u.h[1] = *(const v8h*)(p + 8);
  return u.v;
}

// ---------------------------------------------------------------------------
// Weight packing: fp32 (K x ldw) row-major -> f16 B-fragments.
// frag = kt*NT + nt ; inside: lane*16 + i ; k = kt*32+(lane>>4)*8+(i&7)+(i>>3)*16, n = nt*16+(lane&15)
// ---------------------------------------------------------------------------
__global__ void pack_weights_k(const float* __restrict__ W, _Float16* __restrict__ dst,
                               int K, int NT, int ldw) {
  int total = (K >> 5) * NT * 512;
  for (int idx = blockIdx.x * blockDim.x + threadIdx.x; idx < total;
       idx += gridDim.x * blockDim.x) {
    int frag = idx >> 9;
    int rem  = idx & 511;
    int lane = rem >> 4;
    int i    = rem & 15;
    int kt   = frag / NT;
    int nt   = frag - kt * NT;
    int k = kt * 32 + (lane >> 4) * 8 + (i & 7) + (i >> 3) * 16;
    int n = nt * 16 + (lane & 15);
    float v = (n < ldw) ? W[(size_t)k * ldw + n] : 0.0f;
    dst[idx] = (_Float16)v;
  }
}

// ---------------------------------------------------------------------------
// Degree / norm
// ---------------------------------------------------------------------------
__global__ void set_ones_k(float* __restrict__ p, int n) {
  int i = blockIdx.x * blockDim.x + threadIdx.x;
  if (i < n) p[i] = 1.0f;   // self-loop contribution
}
__global__ void count_col_k(const int* __restrict__ col, float* __restrict__ deg, int ne) {
  int e = blockIdx.x * blockDim.x + threadIdx.x;
  if (e < ne) atomicAdd(&deg[col[e]], 1.0f);
}
__global__ void rsqrt_k(float* __restrict__ p, int n) {
  int i = blockIdx.x * blockDim.x + threadIdx.x;
  if (i < n) p[i] = rsqrtf(p[i]);   // deg >= 1 always
}

// ---------------------------------------------------------------------------
// GCN message passing (EMB = 64)
// ---------------------------------------------------------------------------
__global__ void self_init_k(const float* __restrict__ h, const float* __restrict__ dinv,
                            float* __restrict__ out, int total) {
  int i = blockIdx.x * blockDim.x + threadIdx.x;
  if (i < total) {
    float d = dinv[i >> 6];
    out[i] = h[i] * d * d;
  }
}
__global__ void scatter_k(const int* __restrict__ row, const int* __restrict__ col,
                          const float* __restrict__ dinv, const float* __restrict__ h,
                          float* __restrict__ out, int ne) {
  int e = blockIdx.x * blockDim.x + threadIdx.x;
  if (e >= ne) return;
  int r = row[e], c = col[e];
  float nrm = dinv[r] * dinv[c];
  const float4* hr = (const float4*)(h + (size_t)r * 64);
  float* oc = out + (size_t)c * 64;
#pragma unroll
  for (int q = 0; q < 16; ++q) {
    float4 v = hr[q];
    atomicAdd(oc + 4 * q + 0, v.x * nrm);
    atomicAdd(oc + 4 * q + 1, v.y * nrm);
    atomicAdd(oc + 4 * q + 2, v.z * nrm);
    atomicAdd(oc + 4 * q + 3, v.w * nrm);
  }
}
__global__ void bias_act_k(float* __restrict__ buf, const float* __restrict__ b,
                           int total, int relu) {
  int i = blockIdx.x * blockDim.x + threadIdx.x;
  if (i < total) {
    float v = buf[i] + b[i & 63];
    buf[i] = relu ? fmaxf(v, 0.0f) : v;
  }
}
__global__ void to_half_k(const float* __restrict__ s, _Float16* __restrict__ d, int n) {
  int i = blockIdx.x * blockDim.x + threadIdx.x;
  if (i < n) d[i] = (_Float16)s[i];
}

// ---------------------------------------------------------------------------
// Row-tile GEMM: C(n x 64) = A(n x K) @ B(K x 64), f16 WMMA, no bias/act.
// 8 waves/block, one 16-row tile per wave.
// ---------------------------------------------------------------------------
template <int K>
__global__ __launch_bounds__(256) void gemm_rows_k(const float* __restrict__ A,
                                                   const _Float16* __restrict__ Bp,
                                                   float* __restrict__ C, int nrows) {
  constexpr int KT = K / 32, NT = 4, PITCH = K + 8;
  __shared__ _Float16 tiles[8][16 * PITCH];
  int wave = threadIdx.x >> 5, lane = threadIdx.x & 31;
  int rowbase = (blockIdx.x * 8 + wave) * 16;
  _Float16* my = tiles[wave];
  // stage 16 x K fp32 -> f16 (coalesced float4 loads)
  for (int t = lane; t < 16 * K / 4; t += 32) {
    int r = (t * 4) / K, c = (t * 4) % K;
    int gr = rowbase + r;
    float4 v = make_float4(0.f, 0.f, 0.f, 0.f);
    if (gr < nrows) v = ((const float4*)(A + (size_t)gr * K))[c >> 2];
    _Float16* d = my + r * PITCH + c;
    d[0] = (_Float16)v.x; d[1] = (_Float16)v.y; d[2] = (_Float16)v.z; d[3] = (_Float16)v.w;
  }
  __syncthreads();
  if (rowbase >= nrows) return;
  int n = (lane & 15);
  int mo = (lane >> 4) * 8;
#pragma unroll
  for (int nt = 0; nt < NT; ++nt) {
    v8f acc;
#pragma unroll
    for (int g = 0; g < 8; ++g) acc[g] = 0.0f;
#pragma unroll
    for (int kt = 0; kt < KT; ++kt) {
      v16h a = load_a_frag(my, lane, kt, PITCH);
      v16h b = load_b_frag(Bp + (size_t)(kt * NT + nt) * 512, lane);
      acc = wmma16(a, b, acc);
    }
#pragma unroll
    for (int g = 0; g < 8; ++g) {
      int gr = rowbase + mo + g;
      if (gr < nrows) C[(size_t)gr * 64 + nt * 16 + n] = acc[g];
    }
  }
}

// ---------------------------------------------------------------------------
// Placement head: logits = relu(emb @ Wp1 + bp1) @ Wp2 + bp2   (K=64 -> 64 -> 1)
// ---------------------------------------------------------------------------
__global__ __launch_bounds__(256) void placement_k(const float* __restrict__ emb,
                                                   const _Float16* __restrict__ Wp1p,
                                                   const float* __restrict__ bp1,
                                                   const float* __restrict__ Wp2,
                                                   const float* __restrict__ bp2,
                                                   float* __restrict__ out, int nrows) {
  constexpr int K = 64, PITCH = 72;
  __shared__ _Float16 tiles[8][16 * PITCH];
  int wave = threadIdx.x >> 5, lane = threadIdx.x & 31;
  int rowbase = (blockIdx.x * 8 + wave) * 16;
  _Float16* my = tiles[wave];
  for (int t = lane; t < 16 * K / 4; t += 32) {
    int r = (t * 4) / K, c = (t * 4) % K;
    int gr = rowbase + r;
    float4 v = make_float4(0.f, 0.f, 0.f, 0.f);
    if (gr < nrows) v = ((const float4*)(emb + (size_t)gr * K))[c >> 2];
    _Float16* d = my + r * PITCH + c;
    d[0] = (_Float16)v.x; d[1] = (_Float16)v.y; d[2] = (_Float16)v.z; d[3] = (_Float16)v.w;
  }
  __syncthreads();
  if (rowbase >= nrows) return;
  float part[8];
#pragma unroll
  for (int g = 0; g < 8; ++g) part[g] = 0.0f;
#pragma unroll
  for (int nt = 0; nt < 4; ++nt) {
    v8f acc;
#pragma unroll
    for (int g = 0; g < 8; ++g) acc[g] = 0.0f;
#pragma unroll
    for (int kt = 0; kt < 2; ++kt) {
      v16h a = load_a_frag(my, lane, kt, PITCH);
      v16h b = load_b_frag(Wp1p + (size_t)(kt * 4 + nt) * 512, lane);
      acc = wmma16(a, b, acc);
    }
    int n = nt * 16 + (lane & 15);
    float bb = bp1[n], w2 = Wp2[n];
#pragma unroll
    for (int g = 0; g < 8; ++g) part[g] += fmaxf(acc[g] + bb, 0.0f) * w2;
  }
#pragma unroll
  for (int g = 0; g < 8; ++g) {
    float v = part[g];
    v += __shfl_xor(v, 1); v += __shfl_xor(v, 2);
    v += __shfl_xor(v, 4); v += __shfl_xor(v, 8);
    part[g] = v;
  }
  if ((lane & 15) == 0) {
    int mo = (lane >> 4) * 8;
#pragma unroll
    for (int g = 0; g < 8; ++g) {
      int gr = rowbase + mo + g;
      if (gr < nrows) out[gr] = part[g] + bp2[0];
    }
  }
}

// ---------------------------------------------------------------------------
// Fused action head: edge logits + army logits for 16 actions per wave.
// ---------------------------------------------------------------------------
__global__ __launch_bounds__(128) void action_k(
    const _Float16* __restrict__ embh, const int* __restrict__ ae,
    const int* __restrict__ army,
    const _Float16* __restrict__ We1p, const float* __restrict__ be1,
    const float* __restrict__ We2, const float* __restrict__ be2,
    const _Float16* __restrict__ Wa1p, const float* __restrict__ ba1,
    const _Float16* __restrict__ Wa2p, const float* __restrict__ ba2,
    float* __restrict__ edge_out, float* __restrict__ army_out, int nact) {
  constexpr int PITCH = 136;                     // 272B row pitch, 16B aligned frags
  __shared__ _Float16 ee[4][16 * PITCH];         // edge_embed tile, f16
  __shared__ _Float16 h2[4][16 * PITCH];         // army hidden tile, f16
  __shared__ int srca[4][16];
  int wave = threadIdx.x >> 5, lane = threadIdx.x & 31;
  int ngroups = (nact + 15) >> 4;
  for (int grp = blockIdx.x * 4 + wave; grp < ngroups; grp += gridDim.x * 4) {
    int abase = grp * 16;
    int r = lane & 15;
    int half = lane >> 4;
    int a = abase + r; if (a >= nact) a = nact - 1;
    int src = ae[2 * a], tgt = ae[2 * a + 1];
    int node = half ? tgt : src;
    // stage edge_embed row: lanes 0-15 copy emb[src] -> cols 0..63,
    //                       lanes 16-31 copy emb[tgt] -> cols 64..127
    {
      const uint4* sp = (const uint4*)(embh + (size_t)node * 64);
      uint4* dp = (uint4*)(&ee[wave][r * PITCH + (half << 6)]);
#pragma unroll
      for (int q = 0; q < 8; ++q) dp[q] = sp[q];
    }
    if (!half) srca[wave][r] = army[src];

    // ---- edge MLP: relu(ee @ We1 + be1) . We2 + be2 ----
    float ep[8];
#pragma unroll
    for (int g = 0; g < 8; ++g) ep[g] = 0.0f;
#pragma unroll
    for (int nt = 0; nt < 4; ++nt) {
      v8f acc;
#pragma unroll
      for (int g = 0; g < 8; ++g) acc[g] = 0.0f;
#pragma unroll
      for (int kt = 0; kt < 4; ++kt) {
        v16h A = load_a_frag(&ee[wave][0], lane, kt, PITCH);
        v16h B = load_b_frag(We1p + (size_t)(kt * 4 + nt) * 512, lane);
        acc = wmma16(A, B, acc);
      }
      int n = nt * 16 + (lane & 15);
      float bb = be1[n], w2 = We2[n];
#pragma unroll
      for (int g = 0; g < 8; ++g) ep[g] += fmaxf(acc[g] + bb, 0.0f) * w2;
    }
#pragma unroll
    for (int g = 0; g < 8; ++g) {
      float v = ep[g];
      v += __shfl_xor(v, 1); v += __shfl_xor(v, 2);
      v += __shfl_xor(v, 4); v += __shfl_xor(v, 8);
      ep[g] = v;
    }
    if ((lane & 15) == 0) {
      int mo = half * 8;
#pragma unroll
      for (int g = 0; g < 8; ++g) {
        int aa = abase + mo + g;
        if (aa < nact) {
          int s2 = ae[2 * aa], t2 = ae[2 * aa + 1];
          int sa = army[s2], ta = army[t2];
          float pen = ((sa <= 2) || (ta >= 3 * sa)) ? 1.0f : 0.0f;
          if (s2 == t2) pen += 100.0f;
          edge_out[aa] = ep[g] + be2[0] - pen;
        }
      }
    }

    // ---- army hidden: relu(ee @ Wa1 + ba1) -> f16 LDS restage ----
#pragma unroll
    for (int nt = 0; nt < 8; ++nt) {
      v8f acc;
#pragma unroll
      for (int g = 0; g < 8; ++g) acc[g] = 0.0f;
#pragma unroll
      for (int kt = 0; kt < 4; ++kt) {
        v16h A = load_a_frag(&ee[wave][0], lane, kt, PITCH);
        v16h B = load_b_frag(Wa1p + (size_t)(kt * 8 + nt) * 512, lane);
        acc = wmma16(A, B, acc);
      }
      int n = nt * 16 + (lane & 15);
      float bb = ba1[n];
      int mo = half * 8;
#pragma unroll
      for (int g = 0; g < 8; ++g)
        h2[wave][(mo + g) * PITCH + n] = (_Float16)fmaxf(acc[g] + bb, 0.0f);
    }

    // ---- army out: h2 @ Wa2 + ba2, masked by max_sendable ----
#pragma unroll
    for (int nt = 0; nt < 4; ++nt) {
      v8f acc;
#pragma unroll
      for (int g = 0; g < 8; ++g) acc[g] = 0.0f;
#pragma unroll
      for (int kt = 0; kt < 4; ++kt) {
        v16h A = load_a_frag(&h2[wave][0], lane, kt, PITCH);
        v16h B = load_b_frag(Wa2p + (size_t)(kt * 4 + nt) * 512, lane);
        acc = wmma16(A, B, acc);
      }
      int n = nt * 16 + (lane & 15);
      if (n < 50) {
        float bb = ba2[n];
        int mo = half * 8;
#pragma unroll
        for (int g = 0; g < 8; ++g) {
          int m = mo + g;
          int aa = abase + m;
          if (aa < nact) {
            int maxsend = srca[wave][m] - 1;
            float v = (n <= maxsend) ? (acc[g] + bb) : -1000000000.0f;
            army_out[(size_t)aa * 50 + n] = v;
          }
        }
      }
    }
  }
}

// ---------------------------------------------------------------------------
// Host launcher
// ---------------------------------------------------------------------------
extern "C" void kernel_launch(void* const* d_in, const int* in_sizes, int n_in,
                              void* d_out, int out_size, void* d_ws, size_t ws_size,
                              hipStream_t stream) {
  const float* x    = (const float*)d_in[0];
  const int*   ei   = (const int*)d_in[1];
  const int*   ae   = (const int*)d_in[2];
  const int*   army = (const int*)d_in[3];
  const float* W1  = (const float*)d_in[4];
  const float* b1  = (const float*)d_in[5];
  const float* W2  = (const float*)d_in[6];
  const float* b2  = (const float*)d_in[7];
  const float* Wp1 = (const float*)d_in[8];
  const float* bp1 = (const float*)d_in[9];
  const float* Wp2 = (const float*)d_in[10];
  const float* bp2 = (const float*)d_in[11];
  const float* We1 = (const float*)d_in[12];
  const float* be1 = (const float*)d_in[13];
  const float* We2 = (const float*)d_in[14];
  const float* be2 = (const float*)d_in[15];
  const float* Wa1 = (const float*)d_in[16];
  const float* ba1 = (const float*)d_in[17];
  const float* Wa2 = (const float*)d_in[18];
  const float* ba2 = (const float*)d_in[19];

  int nn = in_sizes[3];          // N_NODES
  int ne = in_sizes[1] / 2;      // N_EDGES
  int na = in_sizes[2] / 2;      // N_ACTIONS
  const int* row = ei;
  const int* col = ei + ne;

  float* pl_out   = (float*)d_out;        // (nn,)
  float* edge_out = pl_out + nn;          // (na,)
  float* army_out = edge_out + na;        // (na,50)

  // workspace carve (all 256B-aligned)
  float* buf0 = (float*)d_ws;                       // nn x 64
  float* buf1 = buf0 + (size_t)nn * 64;             // nn x 64
  float* buf2 = buf1 + (size_t)nn * 64;             // nn x 64 (emb)
  float* degv = buf2 + (size_t)nn * 64;             // nn (deg -> dinv)
  _Float16* embh = (_Float16*)(degv + ((nn + 63) & ~63));
  _Float16* W1p  = embh + (size_t)nn * 64;
  _Float16* W2p  = W1p + 8192;
  _Float16* Wp1p = W2p + 4096;
  _Float16* We1p = Wp1p + 4096;
  _Float16* Wa1p = We1p + 8192;
  _Float16* Wa2p = Wa1p + 16384;

  int total64 = nn * 64;
  dim3 b256(256);

  // pack weights -> f16 B-fragments
  pack_weights_k<<<32, b256, 0, stream>>>(W1,  W1p,  128, 4, 64);
  pack_weights_k<<<32, b256, 0, stream>>>(W2,  W2p,   64, 4, 64);
  pack_weights_k<<<32, b256, 0, stream>>>(Wp1, Wp1p,  64, 4, 64);
  pack_weights_k<<<32, b256, 0, stream>>>(We1, We1p, 128, 4, 64);
  pack_weights_k<<<32, b256, 0, stream>>>(Wa1, Wa1p, 128, 8, 128);
  pack_weights_k<<<32, b256, 0, stream>>>(Wa2, Wa2p, 128, 4, 50);

  // degree -> dinv
  set_ones_k<<<(nn + 255) / 256, b256, 0, stream>>>(degv, nn);
  count_col_k<<<(ne + 255) / 256, b256, 0, stream>>>(col, degv, ne);
  rsqrt_k<<<(nn + 255) / 256, b256, 0, stream>>>(degv, nn);

  int tiles = (nn + 15) / 16;
  int gblocks = (tiles + 7) / 8;

  // layer 1: h = relu(gcn(x, W1) + b1)
  gemm_rows_k<128><<<gblocks, b256, 0, stream>>>(x, W1p, buf0, nn);
  self_init_k<<<(total64 + 255) / 256, b256, 0, stream>>>(buf0, degv, buf1, total64);
  scatter_k<<<(ne + 255) / 256, b256, 0, stream>>>(row, col, degv, buf0, buf1, ne);
  bias_act_k<<<(total64 + 255) / 256, b256, 0, stream>>>(buf1, b1, total64, 1);

  // layer 2: emb = gcn(h, W2) + b2
  gemm_rows_k<64><<<gblocks, b256, 0, stream>>>(buf1, W2p, buf0, nn);
  self_init_k<<<(total64 + 255) / 256, b256, 0, stream>>>(buf0, degv, buf2, total64);
  scatter_k<<<(ne + 255) / 256, b256, 0, stream>>>(row, col, degv, buf0, buf2, ne);
  bias_act_k<<<(total64 + 255) / 256, b256, 0, stream>>>(buf2, b2, total64, 0);

  // placement head
  placement_k<<<gblocks, b256, 0, stream>>>(buf2, Wp1p, bp1, Wp2, bp2, pl_out, nn);

  // emb -> f16 for action gathers
  to_half_k<<<(total64 + 255) / 256, b256, 0, stream>>>(buf2, embh, total64);

  // fused action heads (edge + army logits)
  action_k<<<1024, dim3(128), 0, stream>>>(embh, ae, army,
                                           We1p, be1, We2, be2,
                                           Wa1p, ba1, Wa2p, ba2,
                                           edge_out, army_out, na);
  (void)n_in; (void)out_size; (void)ws_size;
}
